// DiffusionNetBlock_72430328479999
// MI455X (gfx1250) — compile-verified, matrix operand order checked by reading, not run on hardware
//
#include <hip/hip_runtime.h>
#include <hip/hip_bf16.h>

// ---------------------------------------------------------------------------
// DiffusionNet block for MI455X (gfx1250).  f16 WMMA (16x16x32, fp32 accum)
// for all GEMMs; fp32 elementwise + native fp32 global atomics for COO scatter.
// B operands staged in LDS pre-packed in WMMA fragment order (2x ds_load_b128
// per fragment) and fragment loads are software-pipelined so LDS latency
// overlaps WMMA execution instead of serializing on s_wait_dscnt 0.
// ---------------------------------------------------------------------------

#define B_ 4
#define V_ 50000
#define C_ 128
#define K_ 128
#define E_ 400000
#define H_ 128
#define TILES_ ((V_ + 127) / 128)   // 391 row tiles of 128
#define CHUNK_ 256                  // V-chunk for split-K spectral projection
#define PROJ_WGS_ 32                // split-K workgroups per batch

typedef _Float16 half_t;
typedef __attribute__((ext_vector_type(16))) _Float16 v16h;
typedef __attribute__((ext_vector_type(8)))  _Float16 v8h;
typedef __attribute__((ext_vector_type(8)))  float    v8f;

__device__ inline v8f wmma16(v16h a, v16h b, v8f c) {
  // D = A(16x32) * B(32x16) + C, fp32 accumulate
  return __builtin_amdgcn_wmma_f32_16x16x32_f16(false, a, false, b, (short)0, c,
                                                false, false);
}

__device__ inline v8f zero8() {
  v8f z = {0.f, 0.f, 0.f, 0.f, 0.f, 0.f, 0.f, 0.f};
  return z;
}

// A fragment: A[m][k] row-major (lda halves), rows m in [0,16),
// k in [kbase, kbase+32).  Per ISA 7.12.2 a lane's data is two contiguous
// 16-byte runs: [kbase+8*hf, +8) and [kbase+16+8*hf, +8).
__device__ inline v16h load_frag_a(const half_t* A, int lda, int kbase) {
  const int l = threadIdx.x & 31;
  const int m = l & 15, hf = l >> 4;
  const half_t* base = A + m * lda + kbase + 8 * hf;
  v8h lo = *(const v8h*)(base);
  v8h hi = *(const v8h*)(base + 16);
  return __builtin_shufflevector(lo, hi, 0, 1, 2, 3, 4, 5, 6, 7, 8, 9, 10, 11,
                                 12, 13, 14, 15);
}

// Packed-B layout: for a B matrix [Ktot][128], block bi = (k>>5)*8 + (n>>4)
// holds a 32x16 fragment as 32 lanes x 16 contiguous halves (32B per lane).
__device__ inline int packed_b_offset(int k, int n) {
  const int ks  = k >> 5;
  const int kk  = k & 31;
  const int hf  = (kk >> 3) & 1;
  const int p   = ((kk >> 4) << 2) | ((kk & 7) >> 1);
  const int odd = kk & 1;
  const int idx = 2 * p + odd;
  const int lane = hf * 16 + (n & 15);
  const int ct   = n >> 4;
  return (((ks * 8 + ct) * 32) + lane) * 16 + idx;
}

// Consume one packed fragment: 16 contiguous halves per lane (2x b128 loads).
__device__ inline v16h load_frag_b_packed(const half_t* P, int block) {
  const int l = threadIdx.x & 31;
  const half_t* base = P + ((size_t)block * 32 + l) * 16;
  v8h lo = *(const v8h*)(base);
  v8h hi = *(const v8h*)(base + 8);
  return __builtin_shufflevector(lo, hi, 0, 1, 2, 3, 4, 5, 6, 7, 8, 9, 10, 11,
                                 12, 13, 14, 15);
}

// Software-pipelined 16-row x 128-col GEMM strip over K = KTOT.
// Fragment i+1 is loaded into fresh registers before the WMMA consuming
// fragment i, so ds latency overlaps matrix-pipe execution.
template <int KTOT>
__device__ inline void gemm_strip(const half_t* Abase, int lda,
                                  const half_t* P, v8f acc[8]) {
  v16h a  = load_frag_a(Abase, lda, 0);
  v16h bf = load_frag_b_packed(P, 0);
#pragma unroll
  for (int ks = 0; ks < KTOT; ks += 32) {
#pragma unroll
    for (int ct = 0; ct < 8; ++ct) {
      const int blk = (ks >> 5) * 8 + ct;
      v16h a_n = a;
      v16h b_n = bf;
      if (ct == 7) {
        if (ks + 32 < KTOT) {
          a_n = load_frag_a(Abase, lda, ks + 32);
          b_n = load_frag_b_packed(P, blk + 1);
        }
      } else {
        b_n = load_frag_b_packed(P, blk + 1);
      }
      acc[ct] = wmma16(a, bf, acc[ct]);
      a  = a_n;
      bf = b_n;
    }
  }
}

// ---------------------------------------------------------------------------
__global__ __launch_bounds__(256) void zero_kernel(float* p, int n) {
  for (int i = blockIdx.x * blockDim.x + threadIdx.x; i < n;
       i += gridDim.x * blockDim.x)
    p[i] = 0.f;
}

// ---------------------------------------------------------------------------
// S[b,k,c] = sum_v evecs[b,v,k] * mass[b,v] * x_in[b,v,c]   (split-K over V)
__global__ __launch_bounds__(256) void spectral_project_kernel(
    const float* __restrict__ x_in, const float* __restrict__ mass,
    const float* __restrict__ evecs, float* __restrict__ S) {
  extern __shared__ char smem[];
  half_t* ev_t = (half_t*)smem;                       // [128 K][CHUNK_ V] row-major A
  half_t* xmp  = (half_t*)(smem + K_ * CHUNK_ * 2);   // packed B, Ktot=CHUNK_

  const int b    = blockIdx.x / PROJ_WGS_;
  const int wg   = blockIdx.x % PROJ_WGS_;
  const int wave = threadIdx.x >> 5;
  const int l    = threadIdx.x & 31;
  const int hf   = l >> 4;

  v8f acc[8];
#pragma unroll
  for (int ct = 0; ct < 8; ++ct) acc[ct] = zero8();

  const int nchunks = (V_ + CHUNK_ - 1) / CHUNK_;
  for (int ch = wg; ch < nchunks; ch += PROJ_WGS_) {
    const int v0 = ch * CHUNK_;
    for (int idx = threadIdx.x; idx < CHUNK_ * K_; idx += blockDim.x) {
      const int vv = idx >> 7, k = idx & 127;
      const int v = v0 + vv;
      float e = 0.f, xv = 0.f;
      if (v < V_) {
        e  = evecs[((long)b * V_ + v) * K_ + k];
        xv = mass[b * V_ + v] * x_in[((long)b * V_ + v) * C_ + k];
      }
      ev_t[k * CHUNK_ + vv]       = (half_t)e;    // A: [K row][V col]
      xmp[packed_b_offset(vv, k)] = (half_t)xv;   // B: k-dim = vv, n = c
    }
    __syncthreads();
    gemm_strip<CHUNK_>(ev_t + (wave * 16) * CHUNK_, CHUNK_, xmp, acc);
    __syncthreads();
  }

#pragma unroll
  for (int ct = 0; ct < 8; ++ct) {
#pragma unroll
    for (int i = 0; i < 8; ++i) {
      const int krow = wave * 16 + i + 8 * hf;
      const int c    = ct * 16 + (l & 15);
      unsafeAtomicAdd(&S[((long)b * K_ + krow) * C_ + c], acc[ct][i]);
    }
  }
}

// ---------------------------------------------------------------------------
__global__ __launch_bounds__(256) void scale_kernel(
    float* __restrict__ S, const float* __restrict__ evals,
    const float* __restrict__ tvec) {
  const int idx = blockIdx.x * blockDim.x + threadIdx.x;
  if (idx >= B_ * K_ * C_) return;
  const int c = idx & 127, k = (idx >> 7) & 127, b = idx >> 14;
  const float t = fmaxf(tvec[c], 1e-8f);
  S[idx] *= expf(-evals[b * K_ + k] * t);
}

// ---------------------------------------------------------------------------
// x_diffuse[b,v,c] = sum_k evecs[b,v,k] * S[b,k,c]
__global__ __launch_bounds__(256) void unproject_kernel(
    const float* __restrict__ evecs, const float* __restrict__ S,
    float* __restrict__ xd) {
  extern __shared__ char smem[];
  half_t* Aev = (half_t*)smem;                    // [128 rows][128 K] row-major A
  half_t* Bsp = (half_t*)(smem + 128 * 128 * 2);  // packed B, Ktot=128

  const int b = blockIdx.x / TILES_, tile = blockIdx.x % TILES_;
  const int v0 = tile * 128;
  for (int idx = threadIdx.x; idx < 128 * 128; idx += blockDim.x) {
    const int r = idx >> 7, k = idx & 127;
    const int v = v0 + r;
    Aev[idx] = (half_t)((v < V_) ? evecs[((long)b * V_ + v) * K_ + k] : 0.f);
    Bsp[packed_b_offset(r, k)] =
        (half_t)S[((long)b * K_ + r) * C_ + k];   // B k-dim = r, n = c
  }
  __syncthreads();

  const int wave = threadIdx.x >> 5, l = threadIdx.x & 31, hf = l >> 4;
  v8f acc[8];
#pragma unroll
  for (int ct = 0; ct < 8; ++ct) acc[ct] = zero8();
  gemm_strip<128>(Aev + (wave * 16) * 128, 128, Bsp, acc);
#pragma unroll
  for (int ct = 0; ct < 8; ++ct) {
#pragma unroll
    for (int i = 0; i < 8; ++i) {
      const int row = v0 + wave * 16 + i + 8 * hf;
      const int c   = ct * 16 + (l & 15);
      if (row < V_) xd[((long)b * V_ + row) * C_ + c] = acc[ct][i];
    }
  }
}

// ---------------------------------------------------------------------------
// gx[b,row,:] += gradX[e] * xd[b,col,:]  (and gy), wave per edge
__global__ __launch_bounds__(256) void scatter_kernel(
    const int* __restrict__ rows, const int* __restrict__ cols,
    const float* __restrict__ vx, const float* __restrict__ vy,
    const float* __restrict__ xd, float* __restrict__ gx,
    float* __restrict__ gy) {
  const int gwarp = (blockIdx.x * blockDim.x + threadIdx.x) >> 5;
  const int lane  = threadIdx.x & 31;
  const int nw    = (gridDim.x * blockDim.x) >> 5;
  for (int e = gwarp; e < B_ * E_; e += nw) {
    const int b = e / E_, ee = e - b * E_;
    const int r  = rows[b * E_ + ee];
    const int cd = cols[b * E_ + ee];
    const float fx = vx[b * E_ + ee], fy = vy[b * E_ + ee];
    const float* src = xd + ((long)b * V_ + cd) * C_;
    float* dgx = gx + ((long)b * V_ + r) * C_;
    float* dgy = gy + ((long)b * V_ + r) * C_;
#pragma unroll
    for (int c = lane; c < C_; c += 32) {
      const float xv = src[c];
      unsafeAtomicAdd(dgx + c, fx * xv);
      unsafeAtomicAdd(dgy + c, fy * xv);
    }
  }
}

// ---------------------------------------------------------------------------
// xg = tanh(gx*(gx@A_re - gy@A_im) + gy*(gy@A_re + gx@A_im))
__global__ __launch_bounds__(256) void gradfeat_kernel(
    const float* __restrict__ gx, const float* __restrict__ gy,
    const float* __restrict__ A_re, const float* __restrict__ A_im,
    float* __restrict__ xg) {
  extern __shared__ char smem[];
  half_t* gxh  = (half_t*)smem;           // [128][128] row-major A (+elementwise)
  half_t* gyh  = gxh + 128 * 128;
  half_t* Arep = gyh + 128 * 128;         // packed B
  half_t* Aimp = Arep + 128 * 128;        // packed B
  half_t* Ainp = Aimp + 128 * 128;        // packed B of -A_im (no B-negate f16)

  const int b = blockIdx.x / TILES_, tile = blockIdx.x % TILES_;
  const int v0 = tile * 128;
  for (int idx = threadIdx.x; idx < 128 * 128; idx += blockDim.x) {
    const int r = idx >> 7, c = idx & 127;
    const int v = v0 + r;
    gxh[idx] = (half_t)((v < V_) ? gx[((long)b * V_ + v) * C_ + c] : 0.f);
    gyh[idx] = (half_t)((v < V_) ? gy[((long)b * V_ + v) * C_ + c] : 0.f);
    const float ar = A_re[idx], ai = A_im[idx];
    const int po = packed_b_offset(r, c);
    Arep[po] = (half_t)ar;
    Aimp[po] = (half_t)ai;
    Ainp[po] = (half_t)(-ai);
  }
  __syncthreads();

  const int wave = threadIdx.x >> 5, l = threadIdx.x & 31, hf = l >> 4;
  const half_t* Astrip_x = gxh + (wave * 16) * 128;
  const half_t* Astrip_y = gyh + (wave * 16) * 128;

  for (int ct = 0; ct < 8; ++ct) {
    v8f aR = zero8(), aI = zero8();
    // pipelined 5-fragment bundle across the K loop
    v16h ax  = load_frag_a(Astrip_x, 128, 0);
    v16h ay  = load_frag_a(Astrip_y, 128, 0);
    v16h bR  = load_frag_b_packed(Arep, ct);
    v16h bI  = load_frag_b_packed(Aimp, ct);
    v16h bIn = load_frag_b_packed(Ainp, ct);
#pragma unroll
    for (int ks = 0; ks < 4; ++ks) {
      v16h ax_n = ax, ay_n = ay, bR_n = bR, bI_n = bI, bIn_n = bIn;
      if (ks < 3) {
        const int blk = (ks + 1) * 8 + ct;
        ax_n  = load_frag_a(Astrip_x, 128, (ks + 1) * 32);
        ay_n  = load_frag_a(Astrip_y, 128, (ks + 1) * 32);
        bR_n  = load_frag_b_packed(Arep, blk);
        bI_n  = load_frag_b_packed(Aimp, blk);
        bIn_n = load_frag_b_packed(Ainp, blk);
      }
      aR = wmma16(ax, bR, aR);
      aR = wmma16(ay, bIn, aR);
      aI = wmma16(ay, bR, aI);
      aI = wmma16(ax, bI, aI);
      ax = ax_n; ay = ay_n; bR = bR_n; bI = bI_n; bIn = bIn_n;
    }
#pragma unroll
    for (int i = 0; i < 8; ++i) {
      const int mloc = wave * 16 + i + 8 * hf;
      const int n    = ct * 16 + (l & 15);
      const float gxv = (float)gxh[mloc * 128 + n];
      const float gyv = (float)gyh[mloc * 128 + n];
      const float val = tanhf(gxv * aR[i] + gyv * aI[i]);
      const int row = v0 + mloc;
      if (row < V_) xg[((long)b * V_ + row) * C_ + n] = val;
    }
  }
}

// ---------------------------------------------------------------------------
// out = relu([x_in|xd|xg] @ W0 + b0) @ W1 + b1 + x_in    (fused, 256KB LDS)
__global__ __launch_bounds__(256) void mlp_kernel(
    const float* __restrict__ x_in, const float* __restrict__ xd,
    const float* __restrict__ xg, const float* __restrict__ W0,
    const float* __restrict__ b0, const float* __restrict__ W1,
    const float* __restrict__ b1, float* __restrict__ out) {
  extern __shared__ char smem[];
  half_t* feat = (half_t*)smem;        // [128][384] row-major A
  half_t* W0p  = feat + 128 * 384;     // packed B, Ktot=384
  half_t* hbuf = W0p + 384 * 128;      // [128][128] row-major A
  half_t* W1p  = hbuf + 128 * 128;     // packed B, Ktot=128

  const int b = blockIdx.x / TILES_, tile = blockIdx.x % TILES_;
  const int v0 = tile * 128;
  for (int idx = threadIdx.x; idx < 128 * 384; idx += blockDim.x) {
    const int r = idx / 384, f = idx - r * 384;
    const int v = v0 + r;
    float val = 0.f;
    if (v < V_) {
      if (f < 128)      val = x_in[((long)b * V_ + v) * C_ + f];
      else if (f < 256) val = xd[((long)b * V_ + v) * C_ + (f - 128)];
      else              val = xg[((long)b * V_ + v) * C_ + (f - 256)];
    }
    feat[idx] = (half_t)val;
  }
  for (int idx = threadIdx.x; idx < 384 * 128; idx += blockDim.x) {
    const int k = idx >> 7, n = idx & 127;
    W0p[packed_b_offset(k, n)] = (half_t)W0[idx];
  }
  for (int idx = threadIdx.x; idx < 128 * 128; idx += blockDim.x) {
    const int k = idx >> 7, n = idx & 127;
    W1p[packed_b_offset(k, n)] = (half_t)W1[idx];
  }
  __syncthreads();

  const int wave = threadIdx.x >> 5, l = threadIdx.x & 31, hf = l >> 4;

  // layer 0: h = relu(feat @ W0 + b0)
  v8f acc[8];
#pragma unroll
  for (int ct = 0; ct < 8; ++ct) acc[ct] = zero8();
  gemm_strip<384>(feat + (wave * 16) * 384, 384, W0p, acc);
#pragma unroll
  for (int ct = 0; ct < 8; ++ct) {
#pragma unroll
    for (int i = 0; i < 8; ++i) {
      const int mloc = wave * 16 + i + 8 * hf;
      const int n    = ct * 16 + (l & 15);
      const float hv = fmaxf(acc[ct][i] + b0[n], 0.f);
      hbuf[mloc * 128 + n] = (half_t)hv;
    }
  }
  __syncthreads();

  // layer 1: out = h @ W1 + b1 + x_in
#pragma unroll
  for (int ct = 0; ct < 8; ++ct) acc[ct] = zero8();
  gemm_strip<128>(hbuf + (wave * 16) * 128, 128, W1p, acc);
#pragma unroll
  for (int ct = 0; ct < 8; ++ct) {
#pragma unroll
    for (int i = 0; i < 8; ++i) {
      const int mloc = wave * 16 + i + 8 * hf;
      const int n    = ct * 16 + (l & 15);
      const int row  = v0 + mloc;
      if (row < V_)
        out[((long)b * V_ + row) * C_ + n] =
            acc[ct][i] + b1[n] + x_in[((long)b * V_ + row) * C_ + n];
    }
  }
}

// ---------------------------------------------------------------------------
extern "C" void kernel_launch(void* const* d_in, const int* in_sizes, int n_in,
                              void* d_out, int out_size, void* d_ws,
                              size_t ws_size, hipStream_t stream) {
  const float* x_in  = (const float*)d_in[0];
  const float* mass  = (const float*)d_in[1];
  const float* evals = (const float*)d_in[3];
  const float* evecs = (const float*)d_in[4];
  const int*   grows = (const int*)d_in[5];
  const int*   gcols = (const int*)d_in[6];
  const float* gXv   = (const float*)d_in[7];
  const float* gYv   = (const float*)d_in[8];
  const float* tvec  = (const float*)d_in[9];
  const float* A_re  = (const float*)d_in[10];
  const float* A_im  = (const float*)d_in[11];
  const float* W0    = (const float*)d_in[12];
  const float* b0    = (const float*)d_in[13];
  const float* W1    = (const float*)d_in[14];
  const float* b1    = (const float*)d_in[15];
  float* out = (float*)d_out;

  const long BVC = (long)B_ * V_ * C_;   // 25,600,000
  float* S  = (float*)d_ws;              // [B,K,C]
  float* xd = S + (long)B_ * K_ * C_;    // [B,V,C]
  float* gx = xd + BVC;
  float* gy = gx + BVC;
  float* xg = gy + BVC;

  // zero atomic targets
  zero_kernel<<<64, 256, 0, stream>>>(S, B_ * K_ * C_);
  zero_kernel<<<4096, 256, 0, stream>>>(gx, (int)BVC);
  zero_kernel<<<4096, 256, 0, stream>>>(gy, (int)BVC);

  // 1) spectral projection (split-K over V, atomic reduce)
  spectral_project_kernel<<<B_ * PROJ_WGS_, 256, (K_ * CHUNK_ + CHUNK_ * C_) * 2,
                            stream>>>(x_in, mass, evecs, S);
  // 2) spectral decay
  scale_kernel<<<(B_ * K_ * C_ + 255) / 256, 256, 0, stream>>>(S, evals, tvec);
  // 3) unproject to x_diffuse
  unproject_kernel<<<B_ * TILES_, 256, 2 * 128 * 128 * 2, stream>>>(evecs, S, xd);
  // 4) COO scatter for gx / gy
  scatter_kernel<<<2048, 256, 0, stream>>>(grows, gcols, gXv, gYv, xd, gx, gy);
  // 5) gradient rotation features + tanh
  gradfeat_kernel<<<B_ * TILES_, 256, 5 * 128 * 128 * 2, stream>>>(gx, gy, A_re,
                                                                   A_im, xg);
  // 6) fused MLP + residual
  mlp_kernel<<<B_ * TILES_, 256,
               (128 * 384 + 384 * 128 + 128 * 128 + 128 * 128) * 2, stream>>>(
      x_in, xd, xg, W0, b0, W1, b1, out);
}
